// SlidingWindowAttention_4784593567813
// MI455X (gfx1250) — matrix operand order repaired and checked by reading, not compiled, not run on hardware
//
#include <hip/hip_runtime.h>
#include <hip/hip_bf16.h>

// ---------------------------------------------------------------------------
// Sliding-window attention for MI455X (gfx1250), bf16 WMMA everywhere.
// B=2 S=4096 E=512 H=8 hd=64 window=512 stride=256 -> nw=15, out rows 7680/b.
//
// Pipeline:
//   0) elementwise f32->bf16 conversion of activations + weights (one pass,
//      1/sqrt(hd) folded into Wq so GEMM loops are scale-free)
//   1) QKV projection: pure-bf16 GEMM, 32x64 tile/wave, 8 WMMA per K-step
//   2) flash attention per (b, window, head, 16-query tile); shared-max
//      online softmax (4 shfl + 1 corr-exp per 32-key step)
//   3) output projection: bf16 GEMM, 32x64 tile/wave, f32 out
// ---------------------------------------------------------------------------

#define B_   2
#define S_   4096
#define E_   512
#define H_   8
#define HD_  64
#define NW_  15
#define WIN_ 512
#define STR_ 256
#define ROWS_OUT (NW_ * WIN_)      // 7680 rows per batch after windowing

typedef __bf16 bf16_t;
typedef __attribute__((ext_vector_type(16))) __bf16 v16bf;
typedef __attribute__((ext_vector_type(8)))  __bf16 v8bf;
typedef __attribute__((ext_vector_type(8)))  float  v8f;
typedef __attribute__((ext_vector_type(4)))  float  v4f;

#define SHUF16(lo, hi) \
  __builtin_shufflevector(lo, hi, 0,1,2,3,4,5,6,7,8,9,10,11,12,13,14,15)

static __device__ __forceinline__ v8f wmma_bf16(v16bf a, v16bf b, v8f c) {
  // v_wmma_f32_16x16x32_bf16  D = A*B + C
  return __builtin_amdgcn_wmma_f32_16x16x32_bf16(
      false, a, false, b, (short)0, c, false, false);
}

static __device__ __forceinline__ v8f zero8() {
  v8f z;
#pragma unroll
  for (int i = 0; i < 8; ++i) z[i] = 0.0f;
  return z;
}

// ---- A operand (16Mx32K) from row-major [rows, ld] bf16 array --------------
// lane holds row (lane&15); chunks k0+(lane>>4)*8 and k0+16+(lane>>4)*8.
static __device__ __forceinline__ v16bf
load_a_bf16(const bf16_t* p, long ld, int lane, long row, long k0) {
  const bf16_t* r = p + row * ld + k0 + (long)((lane >> 4) << 3);
  v8bf lo = *(const v8bf*)(r);
  v8bf hi = *(const v8bf*)(r + 16);
  return SHUF16(lo, hi);
}

// ---- B operand (32Kx16N) from row-major B^T = [N, ld] bf16 array -----------
// lane holds col (lane&15); chunks k0+(lane>>4)*8 and k0+16+(lane>>4)*8.
static __device__ __forceinline__ v16bf
load_b_bf16(const bf16_t* p, long ld, int lane, long row, long k0) {
  const bf16_t* r = p + row * ld + k0 + (long)((lane >> 4) << 3);
  v8bf lo = *(const v8bf*)(r);
  v8bf hi = *(const v8bf*)(r + 16);
  return SHUF16(lo, hi);
}

// ---------------------------------------------------------------------------
// Kernel 0: elementwise f32 -> bf16 with scale (vectorized x4).
// ---------------------------------------------------------------------------
__global__ __launch_bounds__(256) void
swa_cvt_bf16(const float* __restrict__ src, bf16_t* __restrict__ dst,
             int n4, float scale) {
  const int i = blockIdx.x * blockDim.x + threadIdx.x;
  if (i >= n4) return;
  v4f x = *(const v4f*)(src + (long)i * 4);
  bf16_t o[4];
#pragma unroll
  for (int j = 0; j < 4; ++j) o[j] = (__bf16)(x[j] * scale);
  *(unsigned long long*)(dst + (long)i * 4) = *(const unsigned long long*)o;
}

// ---------------------------------------------------------------------------
// Kernel 1: QKV projection GEMM (all-bf16).  out = X @ W^T (+ bias in epi).
//  p==0 -> qp [B*S, E]   (W already pre-scaled by 1/sqrt(hd))
//  p==1 -> kp [B*S, E]
//  p==2 -> vt [B, E, S]  (transposed for the P@V WMMA B-operand)
// One wave computes a 32x64 tile: 2 A-tiles share each B load.
// ---------------------------------------------------------------------------
__global__ __launch_bounds__(256) void
swa_qkv_gemm(const bf16_t* __restrict__ xb, const bf16_t* __restrict__ wb,
             const float* __restrict__ bias, bf16_t* __restrict__ qp,
             bf16_t* __restrict__ kp, bf16_t* __restrict__ vt) {
  const int lane = threadIdx.x & 31;
  const int wid = blockIdx.x * 8 + (threadIdx.x >> 5);
  // 3 proj * (8192/32 m-tiles) * (512/64 n-groups) = 6144 wave jobs
  const int p   = wid / 2048;
  const int rem = wid - p * 2048;
  const long m0 = (long)(rem >> 3) * 32;   // row in [0, 8192)
  const long n0 = (long)(rem & 7) * 64;    // col in [0, 512)

  const bf16_t* X  = xb + (long)p * B_ * S_ * E_;
  const bf16_t* Wp = wb + (long)p * E_ * E_;
  const float*  bp = bias + (long)p * E_;
  const float bias_scale = (p == 0) ? 0.125f : 1.0f;  // Wq was pre-scaled

  v8f acc[2][4];
#pragma unroll
  for (int g = 0; g < 2; ++g)
#pragma unroll
    for (int s = 0; s < 4; ++s) acc[g][s] = zero8();

  const int colt = lane & 15;
  for (int kk = 0; kk < E_; kk += 32) {
    v16bf a0 = load_a_bf16(X, E_, lane, m0 + colt, kk);
    v16bf a1 = load_a_bf16(X, E_, lane, m0 + 16 + colt, kk);
#pragma unroll
    for (int s = 0; s < 4; ++s) {
      v16bf b = load_b_bf16(Wp, E_, lane, n0 + s * 16 + colt, kk);
      acc[0][s] = wmma_bf16(a0, b, acc[0][s]);
      acc[1][s] = wmma_bf16(a1, b, acc[1][s]);
    }
  }

  const int half = lane >> 4;
#pragma unroll
  for (int s = 0; s < 4; ++s) {
    const long n = n0 + s * 16 + colt;
    const float bv = bp[n] * bias_scale;
#pragma unroll
    for (int g = 0; g < 2; ++g) {
#pragma unroll
      for (int r = 0; r < 8; ++r) {
        const long m = m0 + g * 16 + r + half * 8;   // flat row in [0, B*S)
        const float val = acc[g][s][r] + bv;
        if (p == 0)      qp[m * E_ + n] = (__bf16)val;
        else if (p == 1) kp[m * E_ + n] = (__bf16)val;
        else             vt[((m / S_) * (long)E_ + n) * S_ + (m % S_)] = (__bf16)val;
      }
    }
  }
}

// ---------------------------------------------------------------------------
// Kernel 2: flash attention over one 512-key window.
// One wave = one (b, window, head, 16-query tile).  Per 32-key step:
//   4 WMMA (scores) + shared-max softmax + LDS bounce + 4 WMMA (P@V).
// Row-group stabilizer: rows 0-7 (lanes 0-15) and rows 8-15 (lanes 16-31)
// each share one running max -> any M >= rowmax keeps softmax exact.
// Denominators: per-lane partials in-loop; one 4-shfl reduce per row at end.
// ---------------------------------------------------------------------------
__global__ __launch_bounds__(256) void
swa_attention(const bf16_t* __restrict__ qp, const bf16_t* __restrict__ kp,
              const bf16_t* __restrict__ vt, bf16_t* __restrict__ ctx) {
  __shared__ bf16_t plds[8][16 * 32];   // 1 KB per wave P-tile staging

  const int lane = threadIdx.x & 31;
  const int warp = threadIdx.x >> 5;
  const int wj = blockIdx.x * 8 + warp; // 2*15*8*32 = 7680 wave jobs
  const int b  = wj / 3840;
  int r1 = wj - b * 3840;
  const int n  = r1 / 256;  r1 -= n * 256;
  const int h  = r1 / 32;
  const int qt = r1 & 31;

  const int half = lane >> 4;
  const int colt = lane & 15;

  const bf16_t* qbase = qp + (long)b * S_ * E_ + h * HD_;
  const bf16_t* kbase = kp + (long)b * S_ * E_ + h * HD_;
  const bf16_t* vbase = vt + ((long)b * E_ + h * HD_) * S_;

  // Q tile (already scaled by 1/sqrt(hd)): rows n*256+qt*16+.., hd split 0/32
  const long qrow = (long)n * STR_ + qt * 16 + colt;
  const v16bf aq0 = load_a_bf16(qbase, E_, lane, qrow, 0);
  const v16bf aq1 = load_a_bf16(qbase, E_, lane, qrow, 32);

  v8f O[4];
#pragma unroll
  for (int d = 0; d < 4; ++d) O[d] = zero8();
  float lpart[8];
#pragma unroll
  for (int r = 0; r < 8; ++r) lpart[r] = 0.0f;
  float mrun = -3.0e38f;   // shared running max for this lane's row group

  bf16_t* myLds = &plds[warp][0];

  for (int ks = 0; ks < WIN_ / 32; ++ks) {
    const long skey = (long)n * STR_ + ks * 32;

    // ---- scores: two 16q x 16k tiles, K-dim = hd = 64 (2 WMMA each) ------
    v8f c0 = zero8(), c1 = zero8();
    {
      v16bf b00 = load_b_bf16(kbase, E_, lane, skey + colt, 0);
      v16bf b01 = load_b_bf16(kbase, E_, lane, skey + colt, 32);
      c0 = wmma_bf16(aq0, b00, c0);
      c0 = wmma_bf16(aq1, b01, c0);
      v16bf b10 = load_b_bf16(kbase, E_, lane, skey + 16 + colt, 0);
      v16bf b11 = load_b_bf16(kbase, E_, lane, skey + 16 + colt, 32);
      c1 = wmma_bf16(aq0, b10, c1);
      c1 = wmma_bf16(aq1, b11, c1);
    }

    // ---- shared-max online softmax ---------------------------------------
    float t = fmaxf(c0[0], c1[0]);
#pragma unroll
    for (int r = 1; r < 8; ++r) t = fmaxf(t, fmaxf(c0[r], c1[r]));
    t = fmaxf(t, __shfl_xor(t, 1, 32));
    t = fmaxf(t, __shfl_xor(t, 2, 32));
    t = fmaxf(t, __shfl_xor(t, 4, 32));
    t = fmaxf(t, __shfl_xor(t, 8, 32));
    const float mnew = fmaxf(mrun, t);
    const float corr = __expf(mrun - mnew);   // == 0 on first iteration
    mrun = mnew;
#pragma unroll
    for (int d = 0; d < 4; ++d)
#pragma unroll
      for (int r = 0; r < 8; ++r) O[d][r] *= corr;

#pragma unroll
    for (int r = 0; r < 8; ++r) {
      const float p0 = __expf(c0[r] - mnew);
      const float p1 = __expf(c1[r] - mnew);
      lpart[r] = lpart[r] * corr + (p0 + p1);
      // stage P (C-layout -> row-major 16x32 tile) in LDS for the A-operand
      const int q = r + half * 8;
      myLds[q * 32 + colt]      = (__bf16)p0;
      myLds[q * 32 + colt + 16] = (__bf16)p1;
    }

    // ---- P @ V: A = P from LDS, B = V^T rows (d-major) -------------------
    v16bf ap;
    {
      const bf16_t* r = myLds + colt * 32 + (half << 3);
      v8bf lo = *(const v8bf*)(r);
      v8bf hi = *(const v8bf*)(r + 16);
      ap = SHUF16(lo, hi);
    }
#pragma unroll
    for (int d = 0; d < 4; ++d) {
      v16bf bv = load_b_bf16(vbase + (long)d * 16 * S_, S_, lane, colt, skey);
      O[d] = wmma_bf16(ap, bv, O[d]);
    }
  }

  // ---- final denominators (one 4-shfl reduce per row), store ctx ---------
#pragma unroll
  for (int r = 0; r < 8; ++r) {
    float rs = lpart[r];
    rs += __shfl_xor(rs, 1, 32);
    rs += __shfl_xor(rs, 2, 32);
    rs += __shfl_xor(rs, 4, 32);
    rs += __shfl_xor(rs, 8, 32);
    const float inv = 1.0f / rs;
    const long row = (long)b * ROWS_OUT + (long)n * WIN_ + qt * 16 + r + half * 8;
#pragma unroll
    for (int d = 0; d < 4; ++d) {
      const long col = h * HD_ + d * 16 + colt;
      ctx[row * E_ + col] = (__bf16)(O[d][r] * inv);
    }
  }
}

// ---------------------------------------------------------------------------
// Kernel 3: output projection (all-bf16 operands, f32 result).
// One wave computes a 32x64 tile: 2 A-tiles share each B load.
// ---------------------------------------------------------------------------
__global__ __launch_bounds__(256) void
swa_out_proj(const bf16_t* __restrict__ ctx, const bf16_t* __restrict__ wob,
             const float* __restrict__ bias, float* __restrict__ out) {
  const int lane = threadIdx.x & 31;
  const int wid = blockIdx.x * 8 + (threadIdx.x >> 5);
  // (15360/32) m-tiles * 8 n-groups = 3840 wave jobs
  const long m0 = (long)(wid >> 3) * 32;
  const long n0 = (long)(wid & 7) * 64;

  v8f acc[2][4];
#pragma unroll
  for (int g = 0; g < 2; ++g)
#pragma unroll
    for (int s = 0; s < 4; ++s) acc[g][s] = zero8();

  const int colt = lane & 15;
  for (int kk = 0; kk < E_; kk += 32) {
    v16bf a0 = load_a_bf16(ctx, E_, lane, m0 + colt, kk);
    v16bf a1 = load_a_bf16(ctx, E_, lane, m0 + 16 + colt, kk);
#pragma unroll
    for (int s = 0; s < 4; ++s) {
      v16bf b = load_b_bf16(wob, E_, lane, n0 + s * 16 + colt, kk);
      acc[0][s] = wmma_bf16(a0, b, acc[0][s]);
      acc[1][s] = wmma_bf16(a1, b, acc[1][s]);
    }
  }

  const int half = lane >> 4;
#pragma unroll
  for (int s = 0; s < 4; ++s) {
    const long n = n0 + s * 16 + colt;
    const float bv = bias[n];
#pragma unroll
    for (int g = 0; g < 2; ++g) {
#pragma unroll
      for (int r = 0; r < 8; ++r) {
        const long m = m0 + g * 16 + r + half * 8;
        out[m * E_ + n] = acc[g][s][r] + bv;
      }
    }
  }
}

// ---------------------------------------------------------------------------
extern "C" void kernel_launch(void* const* d_in, const int* in_sizes, int n_in,
                              void* d_out, int out_size, void* d_ws,
                              size_t ws_size, hipStream_t stream) {
  const float* query = (const float*)d_in[0];
  const float* key   = (const float*)d_in[1];
  const float* value = (const float*)d_in[2];
  const float* in_w  = (const float*)d_in[3];
  const float* in_b  = (const float*)d_in[4];
  const float* out_w = (const float*)d_in[5];
  const float* out_b = (const float*)d_in[6];
  float* out = (float*)d_out;

  // workspace (bf16): xb[3] | wb[3] | wob | qp | kp | vt | ctx
  const size_t proj_elems = (size_t)B_ * S_ * E_;          // 4,194,304
  const size_t w_elems    = (size_t)E_ * E_;               //   262,144
  bf16_t* xb  = (bf16_t*)d_ws;                             // 3 * proj_elems
  bf16_t* wb  = xb + 3 * proj_elems;                       // 3 * w_elems
  bf16_t* wob = wb + 3 * w_elems;                          // w_elems
  bf16_t* qp  = wob + w_elems;
  bf16_t* kp  = qp + proj_elems;
  bf16_t* vt  = kp + proj_elems;
  bf16_t* ctx = vt + proj_elems;
  (void)ws_size; (void)in_sizes; (void)n_in; (void)out_size;

  // ---- 0) conversions (vectorized x4) ------------------------------------
  const int n4_act = (int)(proj_elems / 4);                // 1,048,576
  const int n4_w   = (int)(w_elems / 4);                   //    65,536
  swa_cvt_bf16<<<(n4_act + 255) / 256, 256, 0, stream>>>(query, xb, n4_act, 1.0f);
  swa_cvt_bf16<<<(n4_act + 255) / 256, 256, 0, stream>>>(key, xb + proj_elems, n4_act, 1.0f);
  swa_cvt_bf16<<<(n4_act + 255) / 256, 256, 0, stream>>>(value, xb + 2 * proj_elems, n4_act, 1.0f);
  // Wq pre-scaled by 1/sqrt(hd); Wk, Wv unscaled
  swa_cvt_bf16<<<(n4_w + 255) / 256, 256, 0, stream>>>(in_w, wb, n4_w, 0.125f);
  swa_cvt_bf16<<<(2 * n4_w + 255) / 256, 256, 0, stream>>>(in_w + w_elems, wb + w_elems, 2 * n4_w, 1.0f);
  swa_cvt_bf16<<<(n4_w + 255) / 256, 256, 0, stream>>>(out_w, wob, n4_w, 1.0f);

  // ---- 1) QKV projection: 6144 wave jobs / 8 per block --------------------
  swa_qkv_gemm<<<768, 256, 0, stream>>>(xb, wb, in_b, qp, kp, vt);
  // ---- 2) attention: 7680 wave jobs / 8 per block -------------------------
  swa_attention<<<960, 256, 0, stream>>>(qp, kp, vt, ctx);
  // ---- 3) output projection: 3840 wave jobs / 8 per block -----------------
  swa_out_proj<<<480, 256, 0, stream>>>(ctx, wob, out_b, out);
}
